// get_model_35407710388863
// MI455X (gfx1250) — matrix operand order repaired and checked by reading, not compiled, not run on hardware
//
#include <hip/hip_runtime.h>
#include <cmath>

// ---------------------------------------------------------------------------
// MI455X (gfx1250, wave32) implementation of the point-cloud transformer.
// All dense contractions run through one generic batched-GEMM kernel built on
// v_wmma_f32_16x16x32_f16 (f16 A/B converted on load, f32 accumulate).
// Irregular / precision-critical work (kNN, FPS, softmax, LN, BN stats) is f32.
// Workspace requirement: ~470 MB (see bump allocator in kernel_launch).
// ---------------------------------------------------------------------------

typedef __attribute__((ext_vector_type(16))) _Float16 v16h;
typedef __attribute__((ext_vector_type(8)))  float    v8f;
typedef __attribute__((ext_vector_type(4)))  float    f4;

#define EW 256
static inline unsigned ewb(long total) { return (unsigned)((total + EW - 1) / EW); }

// ---------------------------------------------------------------------------
// Generic batched GEMM:  Y[g1][g2][m][n] = alpha * sum_k A[..]*B[..] (+bias[n])
//   A element: Abase + g1*sA1 + g2*sA2 + m*ldA + k
//   B element: Bbase + g1*sB1 + g2*sB2 + n*ldBn + k*ldBk   (covers W^T and V)
//   Y element: Ybase + g1*sY1 + g2*sY2 + m*ldY + n
// One wave per 16x16 tile; 8 waves / block. act: 0=none, 1=relu.
// Fast paths (wave-uniform selection):
//   - vectorized: 16B-aligned + unit-stride-K operands -> 8x global_load_b128
//     per WMMA (4 for A, 4 for B), no masks.
//   - scalar: unconditional b32 loads with unrolled constant offsets.
// Edge path: clamped in-bounds loads + value select (no EXEC manipulation).
// ---------------------------------------------------------------------------
__global__ __launch_bounds__(256)
void k_gemm(const float* __restrict__ A, const float* __restrict__ Bm,
            const float* __restrict__ bias, float* __restrict__ Y,
            int M, int N, int K,
            int ldA, long sA1, long sA2,
            int ldBn, int ldBk, long sB1, long sB2,
            int ldY, long sY1, long sY2,
            int G2, float alpha, int act)
{
    const int lane   = threadIdx.x & 31;
    const int wave   = threadIdx.x >> 5;
    const int tilesN = (N + 15) >> 4;
    const int tilesM = (M + 15) >> 4;
    const int tile   = blockIdx.x * 8 + wave;
    if (tile >= tilesM * tilesN) return;          // wave-uniform exit
    const int tm = tile / tilesN;
    const int tn = tile - tm * tilesN;
    const int g  = blockIdx.y;
    const int g1 = g / G2, g2 = g - g1 * G2;
    const float* __restrict__ Ab = A  + (long)g1 * sA1 + (long)g2 * sA2;
    const float* __restrict__ Bb = Bm + (long)g1 * sB1 + (long)g2 * sB2;

    const int half = lane >> 4;         // 0: lanes 0-15, 1: lanes 16-31
    const int l16  = lane & 15;
    const int mrow = tm * 16 + l16;     // A-matrix row owned by this lane
    const int ncol = tn * 16 + l16;     // B/C column owned by this lane
    const int mrowC = mrow < M ? mrow : M - 1;   // clamped (always in-bounds)
    const int ncolC = ncol < N ? ncol : N - 1;

    // wave-uniform interior test: all 16 rows and 16 cols valid
    const bool interior = (tm * 16 + 16 <= M) && (tn * 16 + 16 <= N);
    const int  kFull    = interior ? (K & ~31) : 0;

    // per-lane base pointers; in-loop offsets are unrolled constants
    const float* __restrict__ pA = Ab + (long)mrowC * ldA + half * 16;
    const float* __restrict__ pB = Bb + (long)ncolC * ldBn + (long)(half * 16) * ldBk;

    // wave-uniform vectorization tests: with ld % 4 == 0 and a 16B-aligned
    // base, every lane's pointer is 16B aligned (lane offsets are mult. of 4).
    const bool vecA = ((ldA & 3) == 0) && ((((unsigned long)(const void*)Ab) & 15) == 0);
    const bool vecB = (ldBk == 1) && ((ldBn & 3) == 0) &&
                      ((((unsigned long)(const void*)Bb) & 15) == 0);

    v8f acc = {0.f,0.f,0.f,0.f,0.f,0.f,0.f,0.f};
    int k0 = 0;
    if (vecA && vecB) {
        const f4* __restrict__ qA = (const f4*)pA;
        const f4* __restrict__ qB = (const f4*)pB;
        for (; k0 < kFull; k0 += 32) {
            __builtin_prefetch(qA + 8, 0, 0);
            __builtin_prefetch(qB + 8, 0, 0);
            f4 a0 = qA[0], a1 = qA[1], a2 = qA[2], a3 = qA[3];   // floats 0-15
            f4 b0 = qB[0], b1 = qB[1], b2 = qB[2], b3 = qB[3];   // floats 0-15
            v16h av, bv;
#pragma unroll
            for (int j = 0; j < 4; ++j) {
                av[j]      = (_Float16)a0[j];
                av[4 + j]  = (_Float16)a1[j];
                av[8 + j]  = (_Float16)a2[j];
                av[12 + j] = (_Float16)a3[j];
                bv[j]      = (_Float16)b0[j];
                bv[4 + j]  = (_Float16)b1[j];
                bv[8 + j]  = (_Float16)b2[j];
                bv[12 + j] = (_Float16)b3[j];
            }
            acc = __builtin_amdgcn_wmma_f32_16x16x32_f16(false, av, false, bv,
                                                         (short)0, acc, false, false);
            qA += 8;  qB += 8;     // 32 floats
        }
    } else {
        for (; k0 < kFull; k0 += 32) {
            __builtin_prefetch(pA + 32, 0, 0);
            __builtin_prefetch(pB + (long)32 * ldBk, 0, 0);
            v16h av, bv;
#pragma unroll
            for (int e = 0; e < 16; ++e) {
                // A k order used here (same as B): K(e) = half*16 + e
                av[e] = (_Float16)pA[e];
                // B k order used here (same as A): lanes 0-15 K=0..15, lanes 16-31 K=16..31
                bv[e] = (_Float16)pB[(long)e * ldBk];
            }
            acc = __builtin_amdgcn_wmma_f32_16x16x32_f16(false, av, false, bv,
                                                         (short)0, acc, false, false);
            pA += 32;
            pB += (long)32 * ldBk;
        }
    }
    for (; k0 < K; k0 += 32) {   // ragged K and/or edge tiles: branch-free selects
        v16h av, bv;
#pragma unroll
        for (int e = 0; e < 16; ++e) {
            int ka  = k0 + half * 16 + e;
            int kac = ka < K ? ka : K - 1;
            float a = Ab[(long)mrowC * ldA + kac];
            av[e] = (_Float16)((mrow < M && ka < K) ? a : 0.f);
            int kb  = k0 + half * 16 + e;
            int kbc = kb < K ? kb : K - 1;
            float b = Bb[(long)ncolC * ldBn + (long)kbc * ldBk];
            bv[e] = (_Float16)((ncol < N && kb < K) ? b : 0.f);
        }
        acc = __builtin_amdgcn_wmma_f32_16x16x32_f16(false, av, false, bv,
                                                     (short)0, acc, false, false);
    }
    float* Yb = Y + (long)g1 * sY1 + (long)g2 * sY2;
#pragma unroll
    for (int v = 0; v < 8; ++v) {
        int mo = tm * 16 + half * 8 + v;   // C/D layout: VGPR v -> M = half*8+v
        if (mo < M && ncol < N) {
            float val = acc[v] * alpha;
            if (bias) val += bias[ncol];
            if (act == 1) val = val > 0.f ? val : 0.f;
            Yb[(long)mo * ldY + ncol] = val;
        }
    }
}

// --------------------------- kNN selection --------------------------------
// Block per query point: full f32 distances into LDS, Ksel iterative min-
// selection (tie-break lower index == jax top_k order on -dist).
__global__ __launch_bounds__(256)
void k_knn(const float* __restrict__ Q, const float* __restrict__ Cand,
           int* __restrict__ out, int NQ, int NC, int Cdim,
           int ldQ, int ldC, int Ksel)
{
    __shared__ float dist[2048];
    __shared__ float qv[128];
    __shared__ float rv[256];
    __shared__ int   ri[256];
    const int b = blockIdx.y, qi = blockIdx.x, tid = threadIdx.x;
    const float* qp = Q + ((long)b * NQ + qi) * ldQ;
    for (int c = tid; c < Cdim; c += 256) qv[c] = qp[c];
    __syncthreads();
    for (int j = tid; j < NC; j += 256) {
        const float* cp = Cand + ((long)b * NC + j) * ldC;
        float s = 0.f;
        for (int c = 0; c < Cdim; ++c) { float d = qv[c] - cp[c]; s += d * d; }
        dist[j] = s;
    }
    __syncthreads();
    for (int t = 0; t < Ksel; ++t) {
        float bv = 3.4e38f; int bi = NC;
        for (int j = tid; j < NC; j += 256)
            if (dist[j] < bv) { bv = dist[j]; bi = j; }
        rv[tid] = bv; ri[tid] = bi;
        __syncthreads();
        for (int s = 128; s > 0; s >>= 1) {
            if (tid < s) {
                if (rv[tid+s] < rv[tid] || (rv[tid+s] == rv[tid] && ri[tid+s] < ri[tid])) {
                    rv[tid] = rv[tid+s]; ri[tid] = ri[tid+s];
                }
            }
            __syncthreads();
        }
        if (tid == 0) { out[((long)b * NQ + qi) * Ksel + t] = ri[0]; dist[ri[0]] = 3.4e38f; }
        __syncthreads();
    }
}

// ------------------------------- FPS ---------------------------------------
__global__ __launch_bounds__(256)
void k_fps(const float* __restrict__ pts, int ldp, int* __restrict__ fidx,
           int Nn, int npoint)
{
    __shared__ float d[2048];
    __shared__ float rv[256];
    __shared__ int   ri[256];
    __shared__ int   far;
    const int b = blockIdx.x, tid = threadIdx.x;
    for (int j = tid; j < Nn; j += 256) d[j] = 1e10f;
    if (tid == 0) far = 0;
    __syncthreads();
    for (int it = 0; it < npoint; ++it) {
        int f = far;
        if (tid == 0) fidx[b * npoint + it] = f;
        const float* cp = pts + ((long)b * Nn + f) * ldp;
        float cx = cp[0], cy = cp[1], cz = cp[2];
        float bv = -1.f; int bi = 0;
        for (int j = tid; j < Nn; j += 256) {
            const float* pp = pts + ((long)b * Nn + j) * ldp;
            float dx = pp[0]-cx, dy = pp[1]-cy, dz = pp[2]-cz;
            float dd = dx*dx + dy*dy + dz*dz;
            float nd = d[j] < dd ? d[j] : dd;
            d[j] = nd;
            if (nd > bv) { bv = nd; bi = j; }
        }
        rv[tid] = bv; ri[tid] = bi;
        __syncthreads();
        for (int s = 128; s > 0; s >>= 1) {
            if (tid < s) {
                if (rv[tid+s] > rv[tid] || (rv[tid+s] == rv[tid] && ri[tid+s] < ri[tid])) {
                    rv[tid] = rv[tid+s]; ri[tid] = ri[tid+s];
                }
            }
            __syncthreads();
        }
        if (tid == 0) far = ri[0];
        __syncthreads();
    }
}

// ------------------------- edge feature build ------------------------------
__global__ void k_edge_feat(const float* __restrict__ H, const int* __restrict__ idx,
                            float* __restrict__ F, int Nn, int Cdim, int ldH,
                            int Ksel, long total)
{
    long i = (long)blockIdx.x * EW + threadIdx.x;
    if (i >= total) return;
    int  c = (int)(i % Cdim);  long r = i / Cdim;
    int  t = (int)(r % Ksel);  long q = r / Ksel;   // q = b*Nn + n
    long b = q / Nn;
    int  nb = idx[q * Ksel + t];
    float ctr = H[q * ldH + c];
    float nbv = H[((long)b * Nn + nb) * (long)ldH + c];
    float* fp = F + (q * Ksel + t) * (long)(2 * Cdim);
    fp[c]        = nbv - ctr;
    fp[Cdim + c] = ctr;
}

// ------------------------- BN stats / apply --------------------------------
__global__ __launch_bounds__(256)
void k_bn_stats(const float* __restrict__ Y, float* __restrict__ mv,
                long rows, int C, int ldY)
{
    __shared__ double ss[256], s2[256];
    const int c = blockIdx.x, tid = threadIdx.x;
    double a = 0.0, b = 0.0;
    for (long r = tid; r < rows; r += 256) {
        double v = (double)Y[r * ldY + c]; a += v; b += v * v;
    }
    ss[tid] = a; s2[tid] = b;
    __syncthreads();
    for (int s = 128; s > 0; s >>= 1) {
        if (tid < s) { ss[tid] += ss[tid+s]; s2[tid] += s2[tid+s]; }
        __syncthreads();
    }
    if (tid == 0) {
        double m = ss[0] / (double)rows;
        double v = s2[0] / (double)rows - m * m;
        mv[c] = (float)m; mv[C + c] = (float)v;
    }
}

__global__ void k_bn_leaky_max(const float* __restrict__ Y, const float* __restrict__ mv,
                               const float* __restrict__ g, const float* __restrict__ be,
                               const float* __restrict__ res, float* __restrict__ out,
                               long RN, int Ksel, int C, int ldRes, int ldOut)
{
    long i = (long)blockIdx.x * EW + threadIdx.x;
    if (i >= RN * C) return;
    int c = (int)(i % C); long r = i / C;
    float m = mv[c], inv = rsqrtf(mv[C + c] + 1e-5f);
    float gg = g[c], bb = be[c];
    const float* yp = Y + r * (long)Ksel * C + c;
    float best = -3.4e38f;
    for (int t = 0; t < Ksel; ++t) {
        float y = (yp[(long)t * C] - m) * inv * gg + bb;
        y = y >= 0.f ? y : 0.2f * y;
        best = best > y ? best : y;
    }
    if (res) best += res[r * ldRes + c];
    out[r * ldOut + c] = best;
}

__global__ void k_bn_act(const float* __restrict__ Y, const float* __restrict__ mv,
                         const float* __restrict__ g, const float* __restrict__ be,
                         float* __restrict__ out, long rows, int C, int act)
{
    long i = (long)blockIdx.x * EW + threadIdx.x;
    if (i >= rows * (long)C) return;
    int c = (int)(i % C);
    float y = (Y[i] - mv[c]) * rsqrtf(mv[C + c] + 1e-5f) * g[c] + be[c];
    if (act == 1) y = y > 0.f ? y : 0.f;
    out[i] = y;
}

// ------------------------- misc elementwise --------------------------------
__global__ void k_gather_rows(const float* __restrict__ src, const int* __restrict__ idx,
                              float* __restrict__ dst, int nIdx, int Cdim,
                              int srcRows, int ldS, int ldD, long total)
{
    long i = (long)blockIdx.x * EW + threadIdx.x;
    if (i >= total) return;
    int c = (int)(i % Cdim); long r = i / Cdim;       // r = b*nIdx + ii
    long b = r / nIdx;
    int  k = idx[r];
    dst[r * ldD + c] = src[((long)b * srcRows + k) * (long)ldS + c];
}

__global__ void k_add2(const float* __restrict__ A, int ldA,
                       const float* __restrict__ Bp, int ldB,
                       float* __restrict__ O, int ldO, long rows, int C)
{
    long i = (long)blockIdx.x * EW + threadIdx.x;
    if (i >= rows * (long)C) return;
    int c = (int)(i % C); long r = i / C;
    float v = A[r * ldA + c];
    if (Bp) v += Bp[r * ldB + c];
    O[r * ldO + c] = v;
}

// V^T per (seq, head): vt[((seq*nh+h)*hd + d)*S + t] = qkv[(seq*S+t)*3C + 2C + h*hd + d]
__global__ void k_tr_v(const float* __restrict__ qkv, float* __restrict__ vt,
                       int S, int C, int nh, long total)
{
    long i = (long)blockIdx.x * EW + threadIdx.x;
    if (i >= total) return;
    const int hd = C / nh;
    int  t  = (int)(i % S);  long r  = i / S;     // r  = (seq*nh+h)*hd + d
    int  d  = (int)(r % hd); long rh = r / hd;    // rh = seq*nh + h
    int  h  = (int)(rh % nh); long seq = rh / nh;
    vt[i] = qkv[((seq * S + t) * 3L) * C + 2L * C + h * hd + d];
}

__global__ __launch_bounds__(256)
void k_softmax(float* __restrict__ S, int Ncol)
{
    __shared__ float red[256];
    float* p = S + (long)blockIdx.x * Ncol;
    const int tid = threadIdx.x;
    float mx = -3.4e38f;
    for (int j = tid; j < Ncol; j += 256) mx = fmaxf(mx, p[j]);
    red[tid] = mx; __syncthreads();
    for (int s = 128; s > 0; s >>= 1) { if (tid < s) red[tid] = fmaxf(red[tid], red[tid+s]); __syncthreads(); }
    mx = red[0]; __syncthreads();
    float sum = 0.f;
    for (int j = tid; j < Ncol; j += 256) { float e = expf(p[j] - mx); p[j] = e; sum += e; }
    red[tid] = sum; __syncthreads();
    for (int s = 128; s > 0; s >>= 1) { if (tid < s) red[tid] += red[tid+s]; __syncthreads(); }
    float inv = 1.f / red[0];
    for (int j = tid; j < Ncol; j += 256) p[j] *= inv;
}

__global__ __launch_bounds__(256)
void k_ln_res(float* __restrict__ x, const float* __restrict__ h,
              const float* __restrict__ g, const float* __restrict__ b, int C)
{
    __shared__ float r1[256], r2[256];
    float* xp = x + (long)blockIdx.x * C;
    const float* hp = h + (long)blockIdx.x * C;
    const int tid = threadIdx.x;
    float s = 0.f, s2 = 0.f;
    for (int c = tid; c < C; c += 256) { float v = xp[c] + hp[c]; s += v; s2 += v * v; }
    r1[tid] = s; r2[tid] = s2; __syncthreads();
    for (int st = 128; st > 0; st >>= 1) {
        if (tid < st) { r1[tid] += r1[tid+st]; r2[tid] += r2[tid+st]; }
        __syncthreads();
    }
    float m = r1[0] / C, var = r2[0] / C - m * m;
    float inv = rsqrtf(var + 1e-5f);
    for (int c = tid; c < C; c += 256) {
        float v = xp[c] + hp[c];
        xp[c] = (v - m) * inv * g[c] + b[c];
    }
}

__global__ void k_max_mid(const float* __restrict__ in, float* __restrict__ out,
                          long R, int Kmid, int C, int ldIn, int ldOut)
{
    long i = (long)blockIdx.x * EW + threadIdx.x;
    if (i >= R * C) return;
    int c = (int)(i % C); long r = i / C;
    const float* p = in + r * (long)Kmid * ldIn + c;
    float best = -3.4e38f;
    for (int t = 0; t < Kmid; ++t) best = fmaxf(best, p[(long)t * ldIn]);
    out[r * ldOut + c] = best;
}

__global__ void k_interp(const float* __restrict__ loc, float* __restrict__ dst,
                         int Nn, int npoint, int C, int ldDst, long total)
{
    long i = (long)blockIdx.x * EW + threadIdx.x;
    if (i >= total) return;
    int c = (int)(i % C); long r = i / C;             // r = b*Nn + n
    long b = r / Nn; int n = (int)(r - b * Nn);
    float pos = (float)n * ((float)(npoint - 1) / (float)(Nn - 1));
    int lo = (int)floorf(pos);
    int hi = lo + 1; if (hi > npoint - 1) hi = npoint - 1;
    float w = pos - (float)lo;
    float v = loc[((long)b * npoint + lo) * C + c] * (1.f - w)
            + loc[((long)b * npoint + hi) * C + c] * w;
    dst[r * ldDst + c] = v;
}

__global__ void k_bcast(const float* __restrict__ ce, float* __restrict__ dst,
                        int Nn, int C, int ldDst, long total)
{
    long i = (long)blockIdx.x * EW + threadIdx.x;
    if (i >= total) return;
    int c = (int)(i % C); long r = i / C; long b = r / Nn;
    dst[r * ldDst + c] = ce[b * C + c];
}

__global__ void k_tr_in(const float* __restrict__ x, float* __restrict__ pts, long total)
{
    long i = (long)blockIdx.x * EW + threadIdx.x;
    if (i >= total) return;
    int c = (int)(i % 9); long r = i / 9;             // r = b*2048 + n
    int n = (int)(r % 2048); long b = r / 2048;
    pts[i] = x[(b * 9 + c) * 2048 + n];
}

__global__ void k_tr_out(const float* __restrict__ t, float* __restrict__ out, long total)
{
    long i = (long)blockIdx.x * EW + threadIdx.x;
    if (i >= total) return;
    int n = (int)(i % 2048); long r = i / 2048;       // r = b*2 + c
    int c = (int)(r % 2);   long b = r / 2;
    out[i] = t[(b * 2048 + n) * 2 + c];
}

// --------------------------- host helpers ----------------------------------
static inline void gemm(hipStream_t st, const float* A, const float* B, const float* bias,
                        float* Y, int M, int N, int K,
                        int ldA, long sA1, long sA2,
                        int ldBn, int ldBk, long sB1, long sB2,
                        int ldY, long sY1, long sY2,
                        int G1, int G2, float alpha, int act)
{
    int tiles = ((M + 15) / 16) * ((N + 15) / 16);
    dim3 grid((unsigned)((tiles + 7) / 8), (unsigned)(G1 * G2));
    k_gemm<<<grid, 256, 0, st>>>(A, B, bias, Y, M, N, K, ldA, sA1, sA2,
                                 ldBn, ldBk, sB1, sB2, ldY, sY1, sY2, G2, alpha, act);
}
static inline void gemm_nt(hipStream_t st, const float* A, const float* W, const float* bias,
                           float* Y, int M, int N, int K, int ldA, int ldY, int act)
{
    gemm(st, A, W, bias, Y, M, N, K, ldA, 0, 0, K, 1, 0, 0, ldY, 0, 0, 1, 1, 1.0f, act);
}

struct LayerP {
    const float *ff1_b, *ff1_w, *ff2_b, *ff2_w, *ln1_b, *ln1_g, *ln2_b, *ln2_g,
                *o_b, *o_w, *qkv_b, *qkv_w;
};

static void run_encoder(hipStream_t st, float* x, int seqs, int S, int C, int nh,
                        const LayerP* L, int nlayers,
                        float* qkvB, float* scB, float* atB, float* ffB, float* tmpB,
                        float* vtB)
{
    const int hd = C / nh;
    const long rows = (long)seqs * S;
    const float alpha = 1.0f / sqrtf((float)hd);
    for (int l = 0; l < nlayers; ++l) {
        const LayerP& P = L[l];
        gemm_nt(st, x, P.qkv_w, P.qkv_b, qkvB, (int)rows, 3*C, C, C, 3*C, 0);
        // scores[g1=seq][g2=head] = Q @ K^T * alpha
        gemm(st, qkvB, qkvB + C, nullptr, scB, S, S, hd,
             3*C, (long)S*3*C, hd,
             3*C, 1, (long)S*3*C, hd,
             S, (long)nh*S*S, (long)S*S,
             seqs, nh, alpha, 0);
        k_softmax<<<dim3((unsigned)((long)seqs * nh * S)), 256, 0, st>>>(scB, S);
        // V^T so the AV GEMM takes the unit-stride-K vectorized path
        k_tr_v<<<ewb(rows * C), EW, 0, st>>>(qkvB, vtB, S, C, nh, rows * C);
        // attn = probs @ V   (B = V^T: ldBn = S, ldBk = 1)
        gemm(st, scB, vtB, nullptr, atB, S, hd, S,
             S, (long)nh*S*S, (long)S*S,
             S, 1, (long)nh*hd*S, (long)hd*S,
             C, (long)S*C, hd,
             seqs, nh, 1.0f, 0);
        gemm_nt(st, atB, P.o_w, P.o_b, tmpB, (int)rows, C, C, C, C, 0);
        k_ln_res<<<dim3((unsigned)rows), 256, 0, st>>>(x, tmpB, P.ln1_g, P.ln1_b, C);
        gemm_nt(st, x, P.ff1_w, P.ff1_b, ffB, (int)rows, 2*C, C, C, 2*C, 1);
        gemm_nt(st, ffB, P.ff2_w, P.ff2_b, tmpB, (int)rows, C, 2*C, 2*C, C, 0);
        k_ln_res<<<dim3((unsigned)rows), 256, 0, st>>>(x, tmpB, P.ln2_g, P.ln2_b, C);
    }
}

// Param indices: x first, then params flattened in jax pytree (sorted-key) order.
enum {
    IN_X = 0,
    P_DG_B1 = 1, P_DG_B2, P_DG_B3, P_DG_B4,
    P_DG_G1, P_DG_G2, P_DG_G3, P_DG_G4,
    P_DG_W1, P_DG_W2, P_DG_W3, P_DG_W4, P_DG_W5,
    P_GCM0_B, P_GCM0_BE, P_GCM0_G, P_GCM0_W,
    P_GCM1_B, P_GCM1_BE, P_GCM1_G, P_GCM1_W,
    P_GCM2_B, P_GCM2_BE, P_GCM2_G, P_GCM2_W,
    P_GE_B, P_GE_BE, P_GE_G, P_GE_W,
    P_GPE_B1, P_GPE_B2, P_GPE_BE, P_GPE_G, P_GPE_W1, P_GPE_W2,
    P_GT_FC_B, P_GT_FC_W,
    P_GT_L0 = 38,            // 4 layers x 12 entries: 38..85
    P_LPE_B1 = 86, P_LPE_B2, P_LPE_BE, P_LPE_G, P_LPE_W1, P_LPE_W2,
    P_LT_FC_B = 92, P_LT_FC_W,
    P_LT_L0 = 94,            // 94..141
    P_PH_B1 = 142, P_PH_B2, P_PH_BE, P_PH_G, P_PH_W1, P_PH_W2
};

extern "C" void kernel_launch(void* const* d_in, const int* in_sizes, int n_in,
                              void* d_out, int out_size, void* d_ws, size_t ws_size,
                              hipStream_t stream)
{
    (void)in_sizes; (void)n_in; (void)out_size; (void)ws_size;
    auto P = [&](int i) { return (const float*)d_in[i]; };

    const int Bb = 2, Nn = 2048, NP = 512, NS = 32, KDG = 60;
    const long BN = (long)Bb * Nn;           // 4096

    // ---- workspace bump allocator (floats; every size is a multiple of 4,
    //      keeping all buffers 16B-aligned for the vectorized GEMM path) ----
    float* Wf = (float*)d_ws;
    size_t off = 0;
    auto alloc = [&](size_t n) { float* p = Wf + off; off += n; return p; };
    float* pts    = alloc((size_t)BN * 9);
    float* full   = alloc((size_t)BN * 576);       // [dg | locN | g | ce]
    float* xcat   = alloc((size_t)BN * 128);       // [x1 | x2 | x3 | x4]
    float* tLoc   = alloc((size_t)32768 * 128);    // local tokens
    float* tGlb   = alloc((size_t)BN * 256);       // global tokens
    float* maxB   = alloc((size_t)1024 * 128);
    float* locB   = alloc((size_t)1024 * 128);
    float* nxyz   = alloc((size_t)Bb * NP * 3);
    float* gxyz   = alloc((size_t)Bb * NP * NS * 3);
    float* pe1    = alloc((size_t)32768 * 64);
    float* peB    = alloc((size_t)32768 * 128);
    float* gpe1   = alloc((size_t)BN * 128);
    float* gpeB   = alloc((size_t)BN * 256);
    float* qkvB   = alloc((size_t)32768 * 384);
    float* scB    = alloc((size_t)16 * 2048 * 2048);  // also reused for edge scratch
    float* atB    = alloc((size_t)32768 * 128);
    float* ffB    = alloc((size_t)32768 * 256);
    float* tmpB   = alloc((size_t)32768 * 128);
    float* vtB    = alloc((size_t)32768 * 128);    // per-head V^T
    float* yGcm   = alloc((size_t)BN * 256);
    float* hPh    = alloc((size_t)BN * 256);
    float* oPh    = alloc((size_t)BN * 2);
    float* ctx    = alloc((size_t)Bb * 448);
    float* ceB    = alloc((size_t)Bb * 64);
    float* statsB = alloc((size_t)2 * 576);
    int*   knnIdx = (int*)alloc((size_t)BN * KDG);
    int*   fpsIdx = (int*)alloc((size_t)Bb * NP);
    int*   grpIdx = (int*)alloc((size_t)Bb * NP * NS);
    float* edgeF  = scB;                               // lifetimes don't overlap
    float* edgeY  = scB + (size_t)BN * KDG * 128;

    // ---- stage A: pts = x.transpose(0,2,1) ----
    k_tr_in<<<ewb(BN * 9), EW, 0, stream>>>(P(IN_X), pts, BN * 9);

    // ---- stage B: 4 DGCNN edge convs ----
    const float* dgW[4] = {P(P_DG_W1), P(P_DG_W2), P(P_DG_W3), P(P_DG_W4)};
    const float* dgG[4] = {P(P_DG_G1), P(P_DG_G2), P(P_DG_G3), P(P_DG_G4)};
    const float* dgB[4] = {P(P_DG_B1), P(P_DG_B2), P(P_DG_B3), P(P_DG_B4)};
    const float* Hl = pts; int ldH = 9, Cin = 9;
    for (int l = 0; l < 4; ++l) {
        k_knn<<<dim3(Nn, Bb), 256, 0, stream>>>(Hl, Hl, knnIdx, Nn, Nn, Cin, ldH, ldH, KDG);
        long totF = BN * KDG * (long)Cin;
        k_edge_feat<<<ewb(totF), EW, 0, stream>>>(Hl, knnIdx, edgeF, Nn, Cin, ldH, KDG, totF);
        const int Mrows = (int)(BN * KDG);
        gemm_nt(stream, edgeF, dgW[l], nullptr, edgeY, Mrows, 32, 2 * Cin, 2 * Cin, 32, 0);
        k_bn_stats<<<32, 256, 0, stream>>>(edgeY, statsB, Mrows, 32, 32);
        const float* res = (l == 0) ? nullptr : (xcat + (l - 1) * 32);
        k_bn_leaky_max<<<ewb(BN * 32), EW, 0, stream>>>(edgeY, statsB, dgG[l], dgB[l],
                                                        res, xcat + l * 32, BN, KDG, 32, 128, 128);
        Hl = xcat + l * 32; ldH = 128; Cin = 32;
    }
    // dg = concat(x1..x4) @ dg_w5^T  -> full[:, :, 0:128]
    gemm_nt(stream, xcat, P(P_DG_W5), nullptr, full, (int)BN, 128, 128, 128, 576, 0);

    // ---- stage C: FPS + grouping + local PE ----
    k_fps<<<Bb, 256, 0, stream>>>(pts, 9, fpsIdx, Nn, NP);
    k_gather_rows<<<ewb((long)Bb * NP * 3), EW, 0, stream>>>(pts, fpsIdx, nxyz, NP, 3, Nn, 9, 3,
                                                             (long)Bb * NP * 3);
    k_knn<<<dim3(NP, Bb), 256, 0, stream>>>(nxyz, pts, grpIdx, NP, Nn, 3, 3, 9, NS);
    k_gather_rows<<<ewb((long)Bb * NP * NS * 3), EW, 0, stream>>>(pts, grpIdx, gxyz, NP * NS, 3,
                                                                  Nn, 9, 3, (long)Bb * NP * NS * 3);
    k_gather_rows<<<ewb((long)Bb * NP * NS * 128), EW, 0, stream>>>(full, grpIdx, tLoc, NP * NS, 128,
                                                                    Nn, 576, 128,
                                                                    (long)Bb * NP * NS * 128);
    gemm_nt(stream, gxyz, P(P_LPE_W1), P(P_LPE_B1), pe1, 32768, 64, 3, 3, 64, 0);
    k_bn_stats<<<64, 256, 0, stream>>>(pe1, statsB, 32768, 64, 64);
    k_bn_act<<<ewb(32768L * 64), EW, 0, stream>>>(pe1, statsB, P(P_LPE_G), P(P_LPE_BE), pe1, 32768, 64, 1);
    gemm_nt(stream, pe1, P(P_LPE_W2), P(P_LPE_B2), peB, 32768, 128, 64, 64, 128, 0);
    k_add2<<<ewb(32768L * 128), EW, 0, stream>>>(tLoc, 128, peB, 128, tLoc, 128, 32768, 128);

    // ---- local transformer ----
    LayerP lt[4], gt[4];
    auto mkLayer = [&](int base) {
        LayerP L;
        L.ff1_b = P(base+0); L.ff1_w = P(base+1); L.ff2_b = P(base+2); L.ff2_w = P(base+3);
        L.ln1_b = P(base+4); L.ln1_g = P(base+5); L.ln2_b = P(base+6); L.ln2_g = P(base+7);
        L.o_b   = P(base+8); L.o_w   = P(base+9); L.qkv_b = P(base+10); L.qkv_w = P(base+11);
        return L;
    };
    for (int l = 0; l < 4; ++l) { lt[l] = mkLayer(P_LT_L0 + 12*l); gt[l] = mkLayer(P_GT_L0 + 12*l); }
    run_encoder(stream, tLoc, 1024, NS, 128, 8, lt, 4, qkvB, scB, atB, ffB, tmpB, vtB);

    // loc = max over samples @ lt_fc  -> interp to full[:, :, 128:256]
    k_max_mid<<<ewb(1024L * 128), EW, 0, stream>>>(tLoc, maxB, 1024, NS, 128, 128, 128);
    gemm_nt(stream, maxB, P(P_LT_FC_W), P(P_LT_FC_B), locB, 1024, 128, 128, 128, 128, 0);
    k_interp<<<ewb(BN * 128), EW, 0, stream>>>(locB, full + 128, Nn, NP, 128, 576, BN * 128);

    // ---- global PE + global transformer ----
    gemm_nt(stream, pts, P(P_GPE_W1), P(P_GPE_B1), gpe1, (int)BN, 128, 3, 9, 128, 0);
    k_bn_stats<<<128, 256, 0, stream>>>(gpe1, statsB, BN, 128, 128);
    k_bn_act<<<ewb(BN * 128), EW, 0, stream>>>(gpe1, statsB, P(P_GPE_G), P(P_GPE_BE), gpe1, BN, 128, 1);
    gemm_nt(stream, gpe1, P(P_GPE_W2), P(P_GPE_B2), gpeB, (int)BN, 256, 128, 128, 256, 0);
    k_add2<<<ewb(BN * 256), EW, 0, stream>>>(full, 576, gpeB, 256, tGlb, 256, BN, 256);
    run_encoder(stream, tGlb, Bb, Nn, 256, 8, gt, 4, qkvB, scB, atB, ffB, tmpB, vtB);
    gemm_nt(stream, tGlb, P(P_GT_FC_W), P(P_GT_FC_B), full + 256, (int)BN, 256, 256, 256, 576, 0);

    // ---- context max-pool heads (gcm) + ce ----
    const float* gcmW[3]  = {P(P_GCM0_W),  P(P_GCM1_W),  P(P_GCM2_W)};
    const float* gcmB[3]  = {P(P_GCM0_B),  P(P_GCM1_B),  P(P_GCM2_B)};
    const float* gcmG[3]  = {P(P_GCM0_G),  P(P_GCM1_G),  P(P_GCM2_G)};
    const float* gcmBE[3] = {P(P_GCM0_BE), P(P_GCM1_BE), P(P_GCM2_BE)};
    const int gcmC[3] = {64, 128, 256};
    int colOff = 0;
    for (int q = 0; q < 3; ++q) {
        const int Cq = gcmC[q];
        gemm_nt(stream, full, gcmW[q], gcmB[q], yGcm, (int)BN, Cq, 512, 576, Cq, 0);
        k_bn_stats<<<Cq, 256, 0, stream>>>(yGcm, statsB, BN, Cq, Cq);
        k_bn_act<<<ewb(BN * Cq), EW, 0, stream>>>(yGcm, statsB, gcmG[q], gcmBE[q], yGcm, BN, Cq, 1);
        k_max_mid<<<ewb((long)Bb * Cq), EW, 0, stream>>>(yGcm, ctx + colOff, Bb, Nn, Cq, Cq, 448);
        colOff += Cq;
    }
    gemm_nt(stream, ctx, P(P_GE_W), P(P_GE_B), ceB, Bb, 64, 448, 448, 64, 0);
    k_bn_stats<<<64, 256, 0, stream>>>(ceB, statsB, Bb, 64, 64);
    k_bn_act<<<ewb((long)Bb * 64), EW, 0, stream>>>(ceB, statsB, P(P_GE_G), P(P_GE_BE), ceB, Bb, 64, 1);
    k_bcast<<<ewb(BN * 64), EW, 0, stream>>>(ceB, full + 512, Nn, 64, 576, BN * 64);

    // ---- final head ----
    gemm_nt(stream, full, P(P_PH_W1), P(P_PH_B1), hPh, (int)BN, 256, 576, 576, 256, 0);
    k_bn_stats<<<256, 256, 0, stream>>>(hPh, statsB, BN, 256, 256);
    k_bn_act<<<ewb(BN * 256), EW, 0, stream>>>(hPh, statsB, P(P_PH_G), P(P_PH_BE), hPh, BN, 256, 0);
    gemm_nt(stream, hPh, P(P_PH_W2), P(P_PH_B2), oPh, (int)BN, 2, 256, 256, 2, 0);
    k_tr_out<<<ewb(BN * 2), EW, 0, stream>>>(oPh, (float*)d_out, BN * 2);
}
